// SoftAlignment_22471268893254
// MI455X (gfx1250) — compile-verified, matrix operand order checked by reading, not run on hardware
//
#include <hip/hip_runtime.h>
#include <hip/hip_bf16.h>

// ---- CDNA5 vector types ----
typedef __attribute__((ext_vector_type(16))) __bf16 v16bf;
typedef __attribute__((ext_vector_type(8)))  float  v8f;
typedef __attribute__((ext_vector_type(4)))  float  v4f;
typedef __attribute__((ext_vector_type(4)))  unsigned int v4u;
typedef __attribute__((ext_vector_type(4)))  unsigned int u32x4;
typedef __attribute__((ext_vector_type(4)))  int i32x4;
typedef __attribute__((ext_vector_type(8)))  int i32x8;

static constexpr int LQn = 2048;
static constexpr int LPn = 2048;
static constexpr int Hn  = 1024;

__device__ __forceinline__ unsigned short f2bf(float x) {
    unsigned int u = __builtin_bit_cast(unsigned int, x);
    u += 0x7FFFu + ((u >> 16) & 1u);          // round-to-nearest-even
    return (unsigned short)(u >> 16);
}

// ---------------------------------------------------------------------------
// K0: convert W (f32 [H,H]) -> bf16 row-major
// ---------------------------------------------------------------------------
__global__ __launch_bounds__(256)
void convert_w_kernel(const float* __restrict__ W, unsigned short* __restrict__ Wbf) {
    int i = (blockIdx.x * 256 + threadIdx.x) * 4;
    v4f w = *(const v4f*)(W + i);
    Wbf[i + 0] = f2bf(w[0]);
    Wbf[i + 1] = f2bf(w[1]);
    Wbf[i + 2] = f2bf(w[2]);
    Wbf[i + 3] = f2bf(w[3]);
}

// ---------------------------------------------------------------------------
// K1: Y = relu(X @ W^T + b).  Block = 8 waves, 16 rows x 1024 cols.
// Per 64-wide K chunk: block cooperatively converts the shared f32 A strip to
// bf16 in LDS once, then each wave runs 2 K-steps x 8 N-tiles of WMMA with
// A from LDS and batch-preloaded W tiles (one load clause -> 8 back-to-back
// v_wmma) from global (W is 2MB bf16 -> L2 resident).
// ---------------------------------------------------------------------------
__global__ __launch_bounds__(256)
void proj_relu_kernel(const float* __restrict__ X,
                      const unsigned short* __restrict__ Wbf,
                      const float* __restrict__ bias,
                      unsigned short* __restrict__ Ylin,
                      unsigned short* __restrict__ YlinT,
                      int do_trans)
{
    __shared__ unsigned short a_lds[16 * 64];     // bf16 A chunk (2 KB)

    const int lane  = threadIdx.x & 31;
    const int wave  = threadIdx.x >> 5;
    const int row0  = blockIdx.x * 16;
    const int col0  = wave * 128;
    const int lrow  = lane & 15;
    const int half  = lane >> 4;
    const int mbase = half * 8;

    v8f acc[8];
#pragma unroll
    for (int t = 0; t < 8; ++t) acc[t] = (v8f){0.f,0.f,0.f,0.f,0.f,0.f,0.f,0.f};

    const int cflat = threadIdx.x * 4;
    const int crow  = cflat >> 6;
    const int ccol  = cflat & 63;
    const float* csrc = X + (size_t)(row0 + crow) * Hn + ccol;

    for (int kc = 0; kc < Hn; kc += 64) {
        __syncthreads();
        // cooperative f32 -> bf16 staging of the shared A strip (4 elems/thread)
        {
            __builtin_prefetch((const void*)(csrc + kc + 64), 0, 0);
            v4f x = *(const v4f*)(csrc + kc);
            unsigned short* dst = a_lds + cflat;
            dst[0] = f2bf(x[0]);
            dst[1] = f2bf(x[1]);
            dst[2] = f2bf(x[2]);
            dst[3] = f2bf(x[3]);
        }
        __syncthreads();
#pragma unroll
        for (int ks = 0; ks < 2; ++ks) {
            int kk = kc + ks * 32;
            v16bf a = *(const v16bf*)(a_lds + lrow * 64 + ks * 32 + half * 16);
            v16bf bm[8];
#pragma unroll
            for (int t = 0; t < 8; ++t)
                bm[t] = *(const v16bf*)(Wbf + (size_t)(col0 + t * 16 + lrow) * Hn
                                              + kk + half * 16);
#pragma unroll
            for (int t = 0; t < 8; ++t)
                acc[t] = __builtin_amdgcn_wmma_f32_16x16x32_bf16(
                    false, a, false, bm[t], (short)0, acc[t], false, false);
        }
    }

    const int b  = row0 >> 11;          // 2048 rows per batch
    const int l0 = row0 & (LQn - 1);

#pragma unroll
    for (int t = 0; t < 8; ++t) {
        int col  = col0 + t * 16 + lrow;
        float bv = bias[col];
        unsigned short us[8];
#pragma unroll
        for (int r = 0; r < 8; ++r) {
            float v = fmaxf(acc[t][r] + bv, 0.0f);
            us[r] = f2bf(v);
            Ylin[(size_t)(row0 + mbase + r) * Hn + col] = us[r];
        }
        if (do_trans) {
            v4u pk;
#pragma unroll
            for (int i = 0; i < 4; ++i)
                pk[i] = (unsigned int)us[2 * i] | ((unsigned int)us[2 * i + 1] << 16);
            *(v4u*)(YlinT + ((size_t)(b * Hn + col) * LQn + l0 + mbase)) = pk;
        }
    }
}

// ---------------------------------------------------------------------------
// K2: flash attention: out = softmax(p_lin @ q_lin^T) @ q_lin  (f32 out)
// Block = 8 waves, owns 16 P-rows.  The 16x1024 bf16 p-tile (32 KB) is staged
// into LDS ONCE via the Tensor Data Mover (tensor_load_to_lds + tensorcnt),
// then reused by all 64 q-chunks.  All GEMM phases batch-preload operands so
// each load clause feeds a burst of back-to-back v_wmma ops.
// ---------------------------------------------------------------------------
__global__ __launch_bounds__(256)
void flash_attn_kernel(const unsigned short* __restrict__ Plin,   // [B,LP,H] bf16
                       const unsigned short* __restrict__ Qlin,   // [B,LQ,H] bf16
                       const unsigned short* __restrict__ QlinT,  // [B,H,LQ] bf16
                       float* __restrict__ Out)                   // [B,LP,H] f32
{
    __shared__ unsigned short p_lds[16 * 1024];      // 32 KB p-tile, bf16
    __shared__ float          s_part[8 * 16 * 32];   // per-wave partial scores
    __shared__ unsigned short p_buf[16 * 32];        // bf16 softmaxed P tile
    __shared__ float          m_buf[16], l_buf[16], f_buf[16];

    const int lane  = threadIdx.x & 31;
    const int wave  = threadIdx.x >> 5;
    const int tile  = blockIdx.x;            // 0 .. B*LP/16-1
    const int b     = tile >> 7;             // 128 tiles per batch
    const int p0    = (tile & 127) * 16;
    const int lrow  = lane & 15;
    const int half  = lane >> 4;
    const int mbase = half * 8;
    const int kbase = wave * 128;

    const unsigned short* ptile = Plin + (size_t)(b * LPn + p0) * Hn;

#if __has_builtin(__builtin_amdgcn_tensor_load_to_lds)
    if (wave == 0) {
        // Tensor DMA descriptor: 2D tile, 16 rows x 1024 bf16, row stride 1024.
        unsigned lds_off = (unsigned)(unsigned long long)(size_t)(void*)p_lds;
        unsigned long long ga = (unsigned long long)(size_t)(const void*)ptile;
        u32x4 g0 = { 1u,                                   // count=1 (valid D#)
                     lds_off,                              // lds_addr [63:32]
                     (unsigned)(ga & 0xFFFFFFFFu),         // global_addr lo
                     (unsigned)((ga >> 32) & 0x01FFFFFFu)  // global_addr hi
                         | (2u << 30) };                   // type=2 ("image")
        i32x8 g1 = { (int)(1u << 16),                      // data_size=1 (2B)
                     (int)(((unsigned)Hn & 0xFFFFu) << 16),// tensor_dim0 lo16
                     (int)((((unsigned)Hn >> 16) & 0xFFFFu)
                         | (16u << 16)),                   // dim0 hi | tensor_dim1
                     (int)(((unsigned)Hn & 0xFFFFu) << 16),// tile_dim0 = 1024
                     16,                                   // tile_dim1 = 16
                     Hn,                                    // tensor_dim0_stride
                     0, 0 };
        i32x4 z4 = {0, 0, 0, 0};
#if __has_include(<hip/amd_detail/amd_gfx1250_TDM.h>)
        i32x8 z8 = {0, 0, 0, 0, 0, 0, 0, 0};
        __builtin_amdgcn_tensor_load_to_lds(g0, g1, z4, z4, z8, 0);
#else
        __builtin_amdgcn_tensor_load_to_lds(g0, g1, z4, z4, 0);
#endif
#if __has_builtin(__builtin_amdgcn_s_wait_tensorcnt)
        __builtin_amdgcn_s_wait_tensorcnt(0);
#endif
    }
#else
    // fallback: cooperative 16-byte copies (32 KB / 256 threads = 128 B each)
    for (int i = threadIdx.x; i < 16 * 1024 / 8; i += 256)
        ((v4u*)p_lds)[i] = ((const v4u*)ptile)[i];
#endif

    if (threadIdx.x < 16) { m_buf[threadIdx.x] = -3.0e38f; l_buf[threadIdx.x] = 0.0f; }

    v8f acc[8];
#pragma unroll
    for (int t = 0; t < 8; ++t) acc[t] = (v8f){0.f,0.f,0.f,0.f,0.f,0.f,0.f,0.f};

    for (int j = 0; j < LQn / 32; ++j) {
        __syncthreads();   // p_lds ready (iter 0) / prev Phase B done with s_part

        // ---- Phase A: partial score tile (16x32) over this wave's K range.
        //      Batch-preload 4 A (LDS) + 8 B (global) operands, then 8 WMMAs.
        v16bf aA[4], bQ0[4], bQ1[4];
#pragma unroll
        for (int ks = 0; ks < 4; ++ks) {
            int kk = kbase + ks * 32;
            aA[ks] = *(const v16bf*)(p_lds + lrow * Hn + kk + half * 16);
            const unsigned short* qp =
                Qlin + (size_t)(b * LQn + j * 32 + lrow) * Hn + kk + half * 16;
            bQ0[ks] = *(const v16bf*)(qp);
            bQ1[ks] = *(const v16bf*)(qp + (size_t)16 * Hn);
        }
        v8f c0 = (v8f){0.f,0.f,0.f,0.f,0.f,0.f,0.f,0.f};
        v8f c1 = (v8f){0.f,0.f,0.f,0.f,0.f,0.f,0.f,0.f};
#pragma unroll
        for (int ks = 0; ks < 4; ++ks) {
            c0 = __builtin_amdgcn_wmma_f32_16x16x32_bf16(
                false, aA[ks], false, bQ0[ks], (short)0, c0, false, false);
            c1 = __builtin_amdgcn_wmma_f32_16x16x32_bf16(
                false, aA[ks], false, bQ1[ks], (short)0, c1, false, false);
        }
        float* sp = s_part + wave * 512;
#pragma unroll
        for (int r = 0; r < 8; ++r) {
            sp[(mbase + r) * 32 + lrow]      = c0[r];
            sp[(mbase + r) * 32 + lrow + 16] = c1[r];
        }
        __syncthreads();

        // ---- Phase B: wave0 reduces partials + online softmax (deterministic)
        if (wave == 0) {
#pragma unroll
            for (int r = 0; r < 16; ++r) {
                float s = 0.0f;
#pragma unroll
                for (int w = 0; w < 8; ++w) s += s_part[w * 512 + r * 32 + lane];
                float mx = s;
#pragma unroll
                for (int off = 16; off >= 1; off >>= 1)
                    mx = fmaxf(mx, __shfl_xor(mx, off, 32));
                float mo = m_buf[r];
                float mn = fmaxf(mo, mx);
                float f  = (mo > -1.0e38f) ? __expf(mo - mn) : 0.0f;
                float pv = __expf(s - mn);
                float sm = pv;
#pragma unroll
                for (int off = 16; off >= 1; off >>= 1)
                    sm += __shfl_xor(sm, off, 32);
                p_buf[r * 32 + lane] = f2bf(pv);
                if (lane == 0) {
                    m_buf[r] = mn;
                    l_buf[r] = l_buf[r] * f + sm;
                    f_buf[r] = f;
                }
            }
        }
        __syncthreads();

        // ---- Phase C: batch-preload 8 q_linT B-tiles + P-tile, rescale acc,
        //      then 8 back-to-back WMMAs.
        v16bf bq[8];
#pragma unroll
        for (int t = 0; t < 8; ++t) {
            int h = wave * 128 + t * 16 + lrow;
            bq[t] = *(const v16bf*)(QlinT + (size_t)(b * Hn + h) * LQn
                                            + j * 32 + half * 16);
        }
        v16bf pa = *(const v16bf*)(p_buf + lrow * 32 + half * 16);
#pragma unroll
        for (int r = 0; r < 8; ++r) {
            float fr = f_buf[mbase + r];
#pragma unroll
            for (int t = 0; t < 8; ++t) acc[t][r] *= fr;
        }
#pragma unroll
        for (int t = 0; t < 8; ++t)
            acc[t] = __builtin_amdgcn_wmma_f32_16x16x32_bf16(
                false, pa, false, bq[t], (short)0, acc[t], false, false);
    }

    __syncthreads();
    // ---- finalize: divide by softmax denominator, store f32
#pragma unroll
    for (int r = 0; r < 8; ++r) {
        float inv = 1.0f / l_buf[mbase + r];
#pragma unroll
        for (int t = 0; t < 8; ++t) {
            int h = wave * 128 + t * 16 + lrow;
            Out[(size_t)(b * LPn + p0 + mbase + r) * Hn + h] = acc[t][r] * inv;
        }
    }
}

// ---------------------------------------------------------------------------
extern "C" void kernel_launch(void* const* d_in, const int* in_sizes, int n_in,
                              void* d_out, int out_size, void* d_ws, size_t ws_size,
                              hipStream_t stream) {
    (void)in_sizes; (void)n_in; (void)out_size; (void)ws_size;
    const float* q    = (const float*)d_in[0];
    const float* p    = (const float*)d_in[1];
    const float* W    = (const float*)d_in[2];
    const float* bias = (const float*)d_in[3];
    float* out        = (float*)d_out;

    char* ws = (char*)d_ws;
    unsigned short* Wbf   = (unsigned short*)(ws);                         //   2 MB
    unsigned short* Qlin  = (unsigned short*)(ws + (  2ull << 20));        //  64 MB
    unsigned short* Plin  = (unsigned short*)(ws + ( 66ull << 20));        //  64 MB
    unsigned short* QlinT = (unsigned short*)(ws + (130ull << 20));        //  64 MB

    // K0: W -> bf16  (1M elems, 4 per thread)
    convert_w_kernel<<<dim3(1024), dim3(256), 0, stream>>>(W, Wbf);

    // K1: projections.  B*L/16 = 2048 row tiles each.
    proj_relu_kernel<<<dim3(2048), dim3(256), 0, stream>>>(q, Wbf, bias, Qlin, QlinT, 1);
    proj_relu_kernel<<<dim3(2048), dim3(256), 0, stream>>>(p, Wbf, bias, Plin, QlinT, 0);

    // K2: fused scores+softmax+context.  B * LP/16 = 2048 blocks.
    flash_attn_kernel<<<dim3(2048), dim3(256), 0, stream>>>(Plin, Qlin, QlinT, out);
}
